// ConceptAttentionLayer_31722628448841
// MI455X (gfx1250) — compile-verified, matrix operand order checked by reading, not run on hardware
//
#include <hip/hip_runtime.h>
#include <math.h>
#include <stdint.h>

// ---------------------------------------------------------------------------
// CDNA5 (gfx1250) two-stage bipartite gated attention.
// Dense projections: wave32 WMMA f16 GEMM (v_wmma_f32_16x16x32_f16) with
//   - pre-transposed f16 weights (both LDS operands read contiguously -> b128)
//   - optional fused row-LayerNorm on the A operand
//   - B tile staged by the Tensor Data Mover (tensor_load_to_lds) when available
// Edge softmax/aggregation via atomics.
// ---------------------------------------------------------------------------

typedef _Float16 h16;
typedef __attribute__((ext_vector_type(16))) _Float16 v16h;
typedef __attribute__((ext_vector_type(8)))  float    v8f;
typedef __attribute__((ext_vector_type(4)))  unsigned int u32x4;
typedef __attribute__((ext_vector_type(8)))  int          i32x8;
typedef __attribute__((ext_vector_type(4)))  int          i32x4;

#define LDT 136   // LDS tile row stride in halves (128 + 8 pad = 272B rows)

#define N_UNITS    65536
#define N_CONCEPTS 4096
#define KQ         8
#define E_EDGES    262144
#define HDIM       128

#if defined(__gfx1250__) && __has_builtin(__builtin_amdgcn_tensor_load_to_lds) && \
    __has_builtin(__builtin_amdgcn_s_wait_tensorcnt)
#define USE_TDM 1
#else
#define USE_TDM 0
#endif

// ---------------------------------------------------------------------------
__device__ __forceinline__ void atomicMaxF(float* addr, float val) {
  unsigned int* ua = (unsigned int*)addr;
  unsigned int old = *ua;
  while (__uint_as_float(old) < val) {
    unsigned int assumed = old;
    old = atomicCAS(ua, assumed, __float_as_uint(val));
    if (old == assumed) break;
  }
}

// 16x32 f16 fragment per CDNA5 layout; identical for A (rows=M) and for B when
// the tile is stored transposed (rows=N, contiguous along K).
// lane<16: elems 0..7 = K[k0..k0+7], elems 8..15 = K[k0+16..k0+23]; lane>=16: +8.
__device__ __forceinline__ v16h load_frag(const h16* base, int lane, int k0) {
  const int r  = lane & 15;
  const int ko = ((lane >> 4) << 3) + k0;
  const h16* p = base + r * LDT + ko;
  v16h x;
#pragma unroll
  for (int i = 0; i < 8; ++i) x[i] = p[i];
#pragma unroll
  for (int i = 0; i < 8; ++i) x[8 + i] = p[16 + i];
  return x;
}

// ---------------------------------------------------------------------------
// WMMA GEMM:  C[M,N] (+)= act( A'[M,K] @ W[K,N] + bias )
//   Bt = transposed f16 weights, Bt[n*ldb + k] = W[k][n].
//   A' = LayerNorm(A; lnG,lnB) if lnG != nullptr (requires K==128).
//   flags: bit0 accumulate into C, bit1 relu.
// Block 256 threads (8 wave32). Tile 64x128, BK=128. Each wave: 16x64 strip.
// ---------------------------------------------------------------------------
__global__ void __launch_bounds__(256)
k_gemm(const float* __restrict__ A, const float* __restrict__ lnG,
       const float* __restrict__ lnB, const h16* __restrict__ Bt, int ldb,
       const float* __restrict__ bias, float* __restrict__ C,
       int M, int N, int K, int flags) {
  __shared__ h16 As[64 * LDT];
  __shared__ h16 Bs[128 * LDT];
  const int tid  = threadIdx.x;
  const int wave = tid >> 5;
  const int lane = tid & 31;
  const int row0 = blockIdx.y << 6;
  const int col0 = blockIdx.x << 7;
  const int mi = wave >> 1;        // 0..3 : 16-row tile
  const int nj = (wave & 1) << 2;  // 0 or 4 : four 16-col tiles

  v8f acc[4] = {{}, {}, {}, {}};
  const int nk = K >> 7;

  for (int kc = 0; kc < nk; ++kc) {
#if USE_TDM
    // ---- B tile via Tensor Data Mover: 128 rows x 256B, padded to 272B rows ----
    if (wave == 0) {
      const unsigned lds_off = (unsigned)(uintptr_t)(const void*)&Bs[0];
      const unsigned long long ga =
          (unsigned long long)(uintptr_t)(Bt + (size_t)col0 * ldb + (kc << 7));
      u32x4 g0 = {1u,                                   // count=1 (valid, user mode)
                  lds_off,                              // D#.lds_addr
                  (unsigned)ga,                         // global_addr[31:0]
                  (unsigned)((ga >> 32) & 0x01FFFFFFu) | (2u << 30)};  // [56:32] | type=2
      i32x8 g1 = {0, 0, 0, 0, 0, 0, 0, 0};
      // data_size=1 (2B), pad_enable, pad_interval code 5 (64 DW = 256B),
      // pad_amount code 3 (4 DW = 16B)  -> matches LDT row stride of 272B.
      g1[0] = (1 << 16) | (1 << 20) | (5 << 22) | (3 << 25);
      g1[1] = (ldb & 0xFFFF) << 16;                     // tensor_dim0 lo16
      g1[2] = ((ldb >> 16) & 0xFFFF) | ((N & 0xFFFF) << 16);  // dim0 hi | dim1 lo
      g1[3] = ((N >> 16) & 0xFFFF) | (128 << 16);       // dim1 hi | tile_dim0=128
      g1[4] = 128;                                      // tile_dim1=128 rows
      g1[5] = ldb;                                      // tensor_dim0_stride lo32
      i32x4 z4 = {0, 0, 0, 0};
      i32x8 z8 = {0, 0, 0, 0, 0, 0, 0, 0};
      __builtin_amdgcn_tensor_load_to_lds(g0, g1, z4, z4, z8, 0);
    }
#else
    for (int i = tid; i < 2048; i += 256) {
      const int r = i >> 4, c8 = (i & 15) << 3;
      const uint4 v = *(const uint4*)(Bt + (size_t)(col0 + r) * ldb + (kc << 7) + c8);
      *(uint4*)&Bs[r * LDT + c8] = v;
    }
#endif
    // ---- stage A tile (f32 -> f16), optional fused LayerNorm ----
    if (lnG) {  // K == 128 guaranteed by construction
      for (int r = wave; r < 64; r += 8) {
        const float4 v = ((const float4*)(A + (size_t)(row0 + r) * 128))[lane];
        float s  = v.x + v.y + v.z + v.w;
        float s2 = v.x * v.x + v.y * v.y + v.z * v.z + v.w * v.w;
#pragma unroll
        for (int off = 16; off; off >>= 1) {
          s  += __shfl_xor(s, off);
          s2 += __shfl_xor(s2, off);
        }
        const float mu   = s * 0.0078125f;
        const float rstd = rsqrtf(s2 * 0.0078125f - mu * mu + 1e-5f);
        const float4 g4 = ((const float4*)lnG)[lane];
        const float4 b4 = ((const float4*)lnB)[lane];
        h16* dp = &As[r * LDT + (lane << 2)];
        dp[0] = (h16)((v.x - mu) * rstd * g4.x + b4.x);
        dp[1] = (h16)((v.y - mu) * rstd * g4.y + b4.y);
        dp[2] = (h16)((v.z - mu) * rstd * g4.z + b4.z);
        dp[3] = (h16)((v.w - mu) * rstd * g4.w + b4.w);
      }
    } else {
      for (int i = tid; i < 2048; i += 256) {
        const int r = i >> 5, c4 = (i & 31) << 2;
        const float* sp = A + (size_t)(row0 + r) * K + (kc << 7) + c4;
        if (kc + 1 < nk) __builtin_prefetch(sp + 128, 0, 1);  // global_prefetch_b8
        const float4 v = *(const float4*)sp;
        h16* dp = &As[r * LDT + c4];
        dp[0] = (h16)v.x; dp[1] = (h16)v.y; dp[2] = (h16)v.z; dp[3] = (h16)v.w;
      }
    }
#if USE_TDM
    if (wave == 0) __builtin_amdgcn_s_wait_tensorcnt(0);
#endif
    __syncthreads();

#pragma unroll
    for (int ks = 0; ks < 4; ++ks) {
      const int k0 = ks << 5;
      const v16h af = load_frag(&As[(mi << 4) * LDT], lane, k0);
#pragma unroll
      for (int j = 0; j < 4; ++j) {
        const v16h bf = load_frag(&Bs[((nj + j) << 4) * LDT], lane, k0);
        acc[j] = __builtin_amdgcn_wmma_f32_16x16x32_f16(false, af, false, bf,
                                                        (short)0, acc[j], false, false);
      }
    }
    __syncthreads();
  }

  // ---- epilogue: C/D layout: lanes 0..15 -> rows +0..7, lanes 16..31 -> +8 ----
  const int n  = lane & 15;
  const int mb = (lane >> 4) << 3;
  const int grow = row0 + (mi << 4) + mb;
  const bool accum = flags & 1;
  const bool relu  = flags & 2;
#pragma unroll
  for (int j = 0; j < 4; ++j) {
    const int gc = col0 + ((nj + j) << 4) + n;
    const float bv = bias ? bias[gc] : 0.f;
#pragma unroll
    for (int v = 0; v < 8; ++v) {
      const size_t i0 = (size_t)(grow + v) * N + gc;
      float o = acc[j][v] + bv + (accum ? C[i0] : 0.f);
      if (relu) o = fmaxf(o, 0.f);
      C[i0] = o;
    }
  }
}

// ---------------------------------------------------------------------------
// Weight convert + transpose: Wt[n*K + k] = (f16) W[k*N + n]
__global__ void k_convert_wt(const float* __restrict__ Wf, h16* __restrict__ Wt,
                             int K, int N) {
  const int i = blockIdx.x * 256 + threadIdx.x;
  if (i >= K * N) return;
  const int k = i / N, n = i - k * N;
  Wt[(size_t)n * K + k] = (h16)Wf[i];
}

// Standalone LayerNorm over 128 features (wave per row).
__global__ void __launch_bounds__(256)
k_layernorm(const float* __restrict__ x, const float* __restrict__ g,
            const float* __restrict__ b, float* __restrict__ y, int M) {
  const int wave = threadIdx.x >> 5, lane = threadIdx.x & 31;
  const int row = (blockIdx.x << 3) + wave;
  if (row >= M) return;
  const float4 v = ((const float4*)(x + (size_t)row * 128))[lane];
  float s  = v.x + v.y + v.z + v.w;
  float s2 = v.x * v.x + v.y * v.y + v.z * v.z + v.w * v.w;
#pragma unroll
  for (int off = 16; off; off >>= 1) {
    s  += __shfl_xor(s, off);
    s2 += __shfl_xor(s2, off);
  }
  const float mu   = s * 0.0078125f;
  const float rstd = rsqrtf(s2 * 0.0078125f - mu * mu + 1e-5f);
  const float4 g4 = ((const float4*)g)[lane];
  const float4 b4 = ((const float4*)b)[lane];
  float4 o;
  o.x = (v.x - mu) * rstd * g4.x + b4.x;
  o.y = (v.y - mu) * rstd * g4.y + b4.y;
  o.z = (v.z - mu) * rstd * g4.z + b4.z;
  o.w = (v.w - mu) * rstd * g4.w + b4.w;
  ((float4*)(y + (size_t)row * 128))[lane] = o;
}

__global__ void k_tile_concepts(const float* __restrict__ cq, float* __restrict__ out) {
  const int i = blockIdx.x * 256 + threadIdx.x;
  if (i >= N_CONCEPTS * HDIM) return;
  out[i] = cq[((i >> 7) & (KQ - 1)) * HDIM + (i & 127)];
}

__global__ void k_zero(float* __restrict__ p, int n) {
  const int i = blockIdx.x * 256 + threadIdx.x;
  if (i < n) p[i] = 0.f;
}

__global__ void k_init_seg(float* __restrict__ mx, float* __restrict__ den, int n) {
  const int i = blockIdx.x * 256 + threadIdx.x;
  if (i < n) { mx[i] = -INFINITY; den[i] = 0.f; }
}

__global__ void k_fix_max(float* __restrict__ mx, int n) {
  const int i = blockIdx.x * 256 + threadIdx.x;
  if (i < n) { float v = mx[i]; mx[i] = __builtin_isfinite(v) ? v : 0.f; }
}

// sim[e,h] = 0.25 * dot16(q[dst], k[src] (+ kr[e])) ; atomic segment max
__global__ void k_edge_sim(const float* __restrict__ q, const float* __restrict__ k,
                           const float* __restrict__ kr,
                           const int* __restrict__ src, const int* __restrict__ dst,
                           float* __restrict__ sim, float* __restrict__ mx, int E) {
  const int idx = blockIdx.x * 256 + threadIdx.x;
  if (idx >= E * 8) return;
  const int e = idx >> 3, h = idx & 7;
  const int s = src[e], d = dst[e];
  const float4* qp = (const float4*)(q + (size_t)d * 128 + (h << 4));
  const float4* kp = (const float4*)(k + (size_t)s * 128 + (h << 4));
  float acc = 0.f;
  if (kr) {
    const float4* rp = (const float4*)(kr + (size_t)e * 128 + (h << 4));
#pragma unroll
    for (int i = 0; i < 4; ++i) {
      const float4 a = qp[i], b = kp[i], c = rp[i];
      acc += a.x * (b.x + c.x) + a.y * (b.y + c.y) + a.z * (b.z + c.z) + a.w * (b.w + c.w);
    }
  } else {
#pragma unroll
    for (int i = 0; i < 4; ++i) {
      const float4 a = qp[i], b = kp[i];
      acc += a.x * b.x + a.y * b.y + a.z * b.z + a.w * b.w;
    }
  }
  const float sv = acc * 0.25f;
  sim[idx] = sv;
  atomicMaxF(&mx[d * 8 + h], sv);
}

__global__ void k_edge_exp(const int* __restrict__ dst, float* __restrict__ sim,
                           const float* __restrict__ mx, float* __restrict__ den, int E) {
  const int idx = blockIdx.x * 256 + threadIdx.x;
  if (idx >= E * 8) return;
  const int e = idx >> 3, h = idx & 7;
  const int d = dst[e];
  const float ex = __expf(sim[idx] - mx[d * 8 + h]);
  sim[idx] = ex;
  atomicAdd(&den[d * 8 + h], ex);
}

__global__ void k_edge_agg(const int* __restrict__ src, const int* __restrict__ dst,
                           const float* __restrict__ v, const float* __restrict__ vr,
                           const float* __restrict__ ex, const float* __restrict__ den,
                           float* __restrict__ agg, int E) {
  const int idx = blockIdx.x * 256 + threadIdx.x;
  if (idx >= E * 128) return;
  const int e = idx >> 7, c = idx & 127, h = c >> 4;
  const int s = src[e], d = dst[e];
  const float w = ex[e * 8 + h] / (den[d * 8 + h] + 1e-16f);
  float val = v[(size_t)s * 128 + c];
  if (vr) val += vr[(size_t)e * 128 + c];
  atomicAdd(&agg[(size_t)d * 128 + c], val * w);
}

__global__ void k_gate(const float* __restrict__ agg, const float* __restrict__ gz,
                       const float* __restrict__ st, float* __restrict__ upd, int n) {
  const int i = blockIdx.x * 256 + threadIdx.x;
  if (i >= n) return;
  const float a = agg[i];
  const float g = 1.f / (1.f + __expf(-gz[i]));
  upd[i] = a + g * (st[i] - a);
}

__global__ void k_mask(const unsigned char* __restrict__ nm,
                       const float* __restrict__ feat, float* __restrict__ out) {
  const int i = blockIdx.x * 256 + threadIdx.x;
  if (i >= N_UNITS * HDIM) return;
  if (nm[i >> 7]) out[i] = feat[i];
}

// ---------------------------------------------------------------------------
// host launch
// ---------------------------------------------------------------------------
extern "C" void kernel_launch(void* const* d_in, const int* in_sizes, int n_in,
                              void* d_out, int out_size, void* d_ws, size_t ws_size,
                              hipStream_t stream) {
  // Input flattening assumption: dict insertion order (top-level and nested
  // param dicts as written in setup_inputs / _make_attn_params).
  enum { I_FEAT = 0, I_RS1 = 1, I_POSQ = 2, I_CQ = 3, P1 = 4, P2 = 30,
         I_EI1 = 54, I_EI2 = 55, I_MASK = 56 };
  enum { LN_SRC_G = 0, LN_SRC_B, LN_DST_G, LN_DST_B, WQ, BQ, WK, WV, BV, WS, BS,
         WG, BG, WO, BO, LN_FF_G, LN_FF_B, W1, B1, W2, B2,
         LN_R_G, LN_R_B, WKR, WVR, BVR };            // s1 extras at +21..+25
  const int LN_RQ_G = 21, LN_RQ_B = 22, WQR = 23;    // s2 extras

  auto F = [&](int i) { return (const float*)d_in[i]; };
  const float* feat = F(I_FEAT);
  const int* src1 = (const int*)d_in[I_EI1]; const int* dst1 = src1 + E_EDGES;
  const int* src2 = (const int*)d_in[I_EI2]; const int* dst2 = src2 + E_EDGES;
  const unsigned char* nmask = (const unsigned char*)d_in[I_MASK];
  float* out = (float*)d_out;

  // -------- workspace arena (floats) --------
  const size_t FE  = (size_t)E_EDGES * 128;
  const size_t F64 = (size_t)N_UNITS * 128;
  const size_t FC  = (size_t)N_CONCEPTS * 128;
  const size_t FH1 = (size_t)N_CONCEPTS * 512;
  const size_t ES  = (size_t)E_EDGES * 8;
  const size_t SEG = (size_t)N_UNITS * 8;
  float* w = (float*)d_ws;
  float* kr    = w; w += FE;    // stage2: hidden2 (65536x512 == FE)
  float* vr    = w; w += FE;    // stage2: agg2|gz2|st2|upd2 (4 x F64 == FE)
  float* u0    = w; w += F64;   // k1 / xd2
  float* u1    = w; w += F64;   // v1 / q2
  float* x2    = w; w += F64;
  float* sim   = w; w += ES;
  float* mx    = w; w += SEG;
  float* den   = w; w += SEG;
  float* conc0 = w; w += FC;
  float* xd1   = w; w += FC;
  float* q1    = w; w += FC;    // stage2: k2
  float* agg1  = w; w += FC;    // stage2: v2
  float* gz1   = w; w += FC;
  float* st1   = w; w += FC;
  float* upd1  = w; w += FC;
  float* x1    = w; w += FC;
  float* conc  = w; w += FC;
  float* hid1  = w; w += FH1;

  // -------- transposed f16 weight pool --------
  h16* hp = (h16*)w;
  const size_t W16 = 16384, WGH = 32768, W1H = 65536, W2H = 65536;
  h16 *WqT1 = hp; hp += W16;  h16 *WkT1 = hp; hp += W16;  h16 *WvT1 = hp; hp += W16;
  h16 *WsT1 = hp; hp += W16;  h16 *WoT1 = hp; hp += W16;  h16 *WgT1 = hp; hp += WGH;
  h16 *W1T1 = hp; hp += W1H;  h16 *W2T1 = hp; hp += W2H;
  h16 *WkrT = hp; hp += W16;  h16 *WvrT = hp; hp += W16;
  h16 *WqT2 = hp; hp += W16;  h16 *WkT2 = hp; hp += W16;  h16 *WvT2 = hp; hp += W16;
  h16 *WsT2 = hp; hp += W16;  h16 *WoT2 = hp; hp += W16;  h16 *WgT2 = hp; hp += WGH;
  h16 *W1T2 = hp; hp += W1H;  h16 *W2T2 = hp; hp += W2H;  h16 *WqrT = hp; hp += W16;

  auto blocks = [](long n) { return (int)((n + 255) / 256); };
  auto conv = [&](const float* Wf, h16* Wt, int K, int N) {
    k_convert_wt<<<blocks((long)K * N), 256, 0, stream>>>(Wf, Wt, K, N);
  };
  auto gemm = [&](const float* A, const float* lnG, const float* lnB,
                  const h16* Bt, int ldb, const float* bias, float* C,
                  int M, int N, int K, int flags) {
    dim3 grid(N >> 7, M >> 6);
    k_gemm<<<grid, 256, 0, stream>>>(A, lnG, lnB, Bt, ldb, bias, C, M, N, K, flags);
  };

#define P1F(x) F(P1 + (x))
#define P2F(x) F(P2 + (x))
  // -------- one-time (per call) weight transpose/convert --------
  conv(P1F(WQ), WqT1, 128, 128);  conv(P1F(WK), WkT1, 128, 128);
  conv(P1F(WV), WvT1, 128, 128);  conv(P1F(WS), WsT1, 128, 128);
  conv(P1F(WO), WoT1, 128, 128);  conv(P1F(WG), WgT1, 256, 128);
  conv(P1F(W1), W1T1, 128, 512);  conv(P1F(W2), W2T1, 512, 128);
  conv(P1F(WKR), WkrT, 128, 128); conv(P1F(WVR), WvrT, 128, 128);
  conv(P2F(WQ), WqT2, 128, 128);  conv(P2F(WK), WkT2, 128, 128);
  conv(P2F(WV), WvT2, 128, 128);  conv(P2F(WS), WsT2, 128, 128);
  conv(P2F(WO), WoT2, 128, 128);  conv(P2F(WG), WgT2, 256, 128);
  conv(P2F(W1), W1T2, 128, 512);  conv(P2F(W2), W2T2, 512, 128);
  conv(P2F(WQR), WqrT, 128, 128);

  // ================= Stage 1: units -> concepts =================
  k_tile_concepts<<<blocks(FC), 256, 0, stream>>>(F(I_CQ), conc0);
  k_layernorm<<<N_CONCEPTS / 8, 256, 0, stream>>>(conc0, P1F(LN_DST_G), P1F(LN_DST_B), xd1, N_CONCEPTS);
  gemm(xd1, nullptr, nullptr, WqT1, 128, P1F(BQ), q1, N_CONCEPTS, 128, 128, 0);
  gemm(feat, P1F(LN_SRC_G), P1F(LN_SRC_B), WkT1, 128, nullptr, u0, N_UNITS, 128, 128, 0);   // k1
  gemm(feat, P1F(LN_SRC_G), P1F(LN_SRC_B), WvT1, 128, P1F(BV), u1, N_UNITS, 128, 128, 0);   // v1
  gemm(F(I_RS1), P1F(LN_R_G), P1F(LN_R_B), WkrT, 128, nullptr, kr, E_EDGES, 128, 128, 0);
  gemm(F(I_RS1), P1F(LN_R_G), P1F(LN_R_B), WvrT, 128, P1F(BVR), vr, E_EDGES, 128, 128, 0);

  k_init_seg<<<blocks(N_CONCEPTS * 8), 256, 0, stream>>>(mx, den, N_CONCEPTS * 8);
  k_edge_sim<<<blocks((long)E_EDGES * 8), 256, 0, stream>>>(q1, u0, kr, src1, dst1, sim, mx, E_EDGES);
  k_fix_max<<<blocks(N_CONCEPTS * 8), 256, 0, stream>>>(mx, N_CONCEPTS * 8);
  k_edge_exp<<<blocks((long)E_EDGES * 8), 256, 0, stream>>>(dst1, sim, mx, den, E_EDGES);
  k_zero<<<blocks(FC), 256, 0, stream>>>(agg1, (int)FC);
  k_edge_agg<<<blocks((long)E_EDGES * 128), 256, 0, stream>>>(src1, dst1, u1, vr, sim, den, agg1, E_EDGES);

  gemm(agg1, nullptr, nullptr, WgT1, 256, P1F(BG), gz1, N_CONCEPTS, 128, 128, 0);
  gemm(xd1, nullptr, nullptr, WgT1 + 128, 256, nullptr, gz1, N_CONCEPTS, 128, 128, 1);
  gemm(xd1, nullptr, nullptr, WsT1, 128, P1F(BS), st1, N_CONCEPTS, 128, 128, 0);
  k_gate<<<blocks(FC), 256, 0, stream>>>(agg1, gz1, st1, upd1, (int)FC);
  (void)hipMemcpyAsync(x1, conc0, FC * sizeof(float), hipMemcpyDeviceToDevice, stream);
  gemm(upd1, nullptr, nullptr, WoT1, 128, P1F(BO), x1, N_CONCEPTS, 128, 128, 1);
  gemm(x1, P1F(LN_FF_G), P1F(LN_FF_B), W1T1, 128, P1F(B1), hid1, N_CONCEPTS, 512, 128, 2); // relu
  (void)hipMemcpyAsync(conc, x1, FC * sizeof(float), hipMemcpyDeviceToDevice, stream);
  gemm(hid1, nullptr, nullptr, W2T1, 512, P1F(B2), conc, N_CONCEPTS, 128, 512, 1);

  // ================= Stage 2: concepts -> units =================
  float* xd2 = u0;  float* q2 = u1;
  float* agg2 = vr; float* gz2 = vr + F64; float* st2 = vr + 2 * F64; float* upd2 = vr + 3 * F64;
  float* k2 = q1;   float* v2 = agg1;      float* hid2 = kr;

  k_layernorm<<<N_UNITS / 8, 256, 0, stream>>>(feat, P2F(LN_DST_G), P2F(LN_DST_B), xd2, N_UNITS);
  gemm(xd2, nullptr, nullptr, WqT2, 128, P2F(BQ), q2, N_UNITS, 128, 128, 0);
  gemm(F(I_POSQ), P2F(LN_RQ_G), P2F(LN_RQ_B), WqrT, 128, nullptr, q2, N_UNITS, 128, 128, 1);
  gemm(conc, P2F(LN_SRC_G), P2F(LN_SRC_B), WkT2, 128, nullptr, k2, N_CONCEPTS, 128, 128, 0);
  gemm(conc, P2F(LN_SRC_G), P2F(LN_SRC_B), WvT2, 128, P2F(BV), v2, N_CONCEPTS, 128, 128, 0);

  k_init_seg<<<blocks(N_UNITS * 8), 256, 0, stream>>>(mx, den, N_UNITS * 8);
  k_edge_sim<<<blocks((long)E_EDGES * 8), 256, 0, stream>>>(q2, k2, nullptr, src2, dst2, sim, mx, E_EDGES);
  k_fix_max<<<blocks(N_UNITS * 8), 256, 0, stream>>>(mx, N_UNITS * 8);
  k_edge_exp<<<blocks((long)E_EDGES * 8), 256, 0, stream>>>(dst2, sim, mx, den, E_EDGES);
  k_zero<<<blocks(F64), 256, 0, stream>>>(agg2, (int)F64);
  k_edge_agg<<<blocks((long)E_EDGES * 128), 256, 0, stream>>>(src2, dst2, v2, nullptr, sim, den, agg2, E_EDGES);

  gemm(agg2, nullptr, nullptr, WgT2, 256, P2F(BG), gz2, N_UNITS, 128, 128, 0);
  gemm(xd2, nullptr, nullptr, WgT2 + 128, 256, nullptr, gz2, N_UNITS, 128, 128, 1);
  gemm(xd2, nullptr, nullptr, WsT2, 128, P2F(BS), st2, N_UNITS, 128, 128, 0);
  k_gate<<<blocks(F64), 256, 0, stream>>>(agg2, gz2, st2, upd2, (int)F64);
  (void)hipMemcpyAsync(x2, feat, F64 * sizeof(float), hipMemcpyDeviceToDevice, stream);
  gemm(upd2, nullptr, nullptr, WoT2, 128, P2F(BO), x2, N_UNITS, 128, 128, 1);
  gemm(x2, P2F(LN_FF_G), P2F(LN_FF_B), W1T2, 128, P2F(B1), hid2, N_UNITS, 512, 128, 2);    // relu
  (void)hipMemcpyAsync(out, x2, F64 * sizeof(float), hipMemcpyDeviceToDevice, stream);
  gemm(hid2, nullptr, nullptr, W2T2, 512, P2F(B2), out, N_UNITS, 128, 512, 1);
  k_mask<<<blocks(F64), 256, 0, stream>>>(nmask, feat, out);
#undef P1F
#undef P2F
  (void)in_sizes; (void)n_in; (void)out_size; (void)ws_size;
}
